// ChipPairEncoderAlpha_54116587930135
// MI455X (gfx1250) — matrix operand order repaired and checked by reading, not compiled, not run on hardware
//
#include <hip/hip_runtime.h>
#include <hip/hip_bf16.h>

typedef __attribute__((ext_vector_type(16))) __bf16 v16bf;
typedef __attribute__((ext_vector_type(8)))  float  v8f;

#define LN_EPS 1e-5f
#define ATTN_SCALE 0.35355339059327373f   // 1/sqrt(8)

// A-fragment K index for 16-bit 16x16x32 WMMA:
// lane group g (0/1), element e (0..15): K = 16*g + e
__device__ __forceinline__ int aK(int e, int g) {
    return (g << 4) + e;
}

// ---------------------------------------------------------------------------
// Kernel 1: embed + positional embedding
// msa[b,s,l,c] = signal[b,s,l]*embed_w[c] + embed_b[c] + pos[l,c]
// ---------------------------------------------------------------------------
__global__ __launch_bounds__(256)
void k_embed(const float* __restrict__ c0, const float* __restrict__ c1,
             const float* __restrict__ c2, const float* __restrict__ c3,
             const float* __restrict__ ew, const float* __restrict__ eb,
             const float* __restrict__ pos, float* __restrict__ msa) {
    int idx = blockIdx.x * 256 + threadIdx.x;      // 64*4*64*32 = 524288
    int c = idx & 31, l = (idx >> 5) & 63, s = (idx >> 11) & 3, b = idx >> 13;
    const float* sig = (s == 0) ? c0 : (s == 1) ? c1 : (s == 2) ? c2 : c3;
    float v = sig[b * 64 + l];
    msa[idx] = v * ew[c] + eb[c] + pos[l * 32 + c];
}

// ---------------------------------------------------------------------------
// Kernel 2: row attention (per (b,s) sequence of 64 bins), residual + LN.
// QKV (64x32)@(32x96) and out-proj (64x32)@(32x32) via WMMA bf16.
// ---------------------------------------------------------------------------
__global__ __launch_bounds__(256)
void k_row_attn(float* __restrict__ msa,
                const float* __restrict__ w_in, const float* __restrict__ b_in,
                const float* __restrict__ w_out, const float* __restrict__ b_out,
                const float* __restrict__ ln_g, const float* __restrict__ ln_b) {
    __shared__ float  xs[64][32];
    __shared__ float  qkv[64][96];
    __shared__ float  sc[64][64];
    __shared__ float  os[64][32];
    __shared__ __bf16 win[96 * 32];
    __shared__ __bf16 wot[32 * 32];

    int t = threadIdx.x;
    float* xg = msa + (size_t)blockIdx.x * 64 * 32;

    for (int i = t; i < 64 * 32; i += 256) (&xs[0][0])[i] = xg[i];
    for (int i = t; i < 96 * 32; i += 256) win[i] = (__bf16)w_in[i];
    for (int i = t; i < 32 * 32; i += 256) wot[i] = (__bf16)w_out[i];
    __syncthreads();

    int wave = t >> 5, lane = t & 31, g = lane >> 4, L = lane & 15;

    // ---- QKV projection: 4x6 tiles of 16x16, K=32 (one WMMA each) ----
    for (int q = 0; q < 3; ++q) {
        int tile = wave * 3 + q;
        int mt = tile / 6, nt = tile % 6;
        v16bf a, bm;
        #pragma unroll
        for (int e = 0; e < 16; ++e) {
            a[e]  = (__bf16)xs[mt * 16 + L][aK(e, g)];
            bm[e] = win[(nt * 16 + L) * 32 + (g << 4) + e];
        }
        float bias = b_in[nt * 16 + L];
        v8f acc;
        #pragma unroll
        for (int r = 0; r < 8; ++r) acc[r] = bias;
        acc = __builtin_amdgcn_wmma_f32_16x16x32_bf16(false, a, false, bm,
                                                      (short)0, acc, false, false);
        #pragma unroll
        for (int r = 0; r < 8; ++r) qkv[mt * 16 + r + 8 * g][nt * 16 + L] = acc[r];
    }
    __syncthreads();

    // ---- attention, one head at a time (d=8, L=64) ----
    for (int h = 0; h < 4; ++h) {
        {   // scores: 64x64, 16 per thread
            int i = t >> 2, jb = t & 3;
            #pragma unroll
            for (int e = 0; e < 16; ++e) {
                int j = jb * 16 + e;
                float s = 0.f;
                #pragma unroll
                for (int dd = 0; dd < 8; ++dd)
                    s += qkv[i][h * 8 + dd] * qkv[j][32 + h * 8 + dd];
                sc[i][j] = s * ATTN_SCALE;
            }
        }
        __syncthreads();
        if (t < 64) {   // softmax over row t
            float mx = -1e30f;
            for (int j = 0; j < 64; ++j) mx = fmaxf(mx, sc[t][j]);
            float sum = 0.f;
            for (int j = 0; j < 64; ++j) { float e = __expf(sc[t][j] - mx); sc[t][j] = e; sum += e; }
            float inv = 1.f / sum;
            for (int j = 0; j < 64; ++j) sc[t][j] *= inv;
        }
        __syncthreads();
        {   // attn @ V: 64x8 outputs, 2 per thread
            #pragma unroll
            for (int q = 0; q < 2; ++q) {
                int id = t * 2 + q;
                int i = id >> 3, dd = id & 7;
                float o = 0.f;
                for (int j = 0; j < 64; ++j) o += sc[i][j] * qkv[j][64 + h * 8 + dd];
                os[i][h * 8 + dd] = o;
            }
        }
        __syncthreads();
    }

    // ---- out projection: 4x2 tiles, one per wave; store into sc ----
    {
        int mt = wave >> 1, nt = wave & 1;
        v16bf a, bm;
        #pragma unroll
        for (int e = 0; e < 16; ++e) {
            a[e]  = (__bf16)os[mt * 16 + L][aK(e, g)];
            bm[e] = wot[(nt * 16 + L) * 32 + (g << 4) + e];
        }
        float bias = b_out[nt * 16 + L];
        v8f acc;
        #pragma unroll
        for (int r = 0; r < 8; ++r) acc[r] = bias;
        acc = __builtin_amdgcn_wmma_f32_16x16x32_bf16(false, a, false, bm,
                                                      (short)0, acc, false, false);
        #pragma unroll
        for (int r = 0; r < 8; ++r) sc[mt * 16 + r + 8 * g][nt * 16 + L] = acc[r];
    }
    __syncthreads();

    // ---- residual + LayerNorm over C=32, write back in place ----
    if (t < 64) {
        float v[32], mu = 0.f;
        for (int c = 0; c < 32; ++c) { v[c] = xs[t][c] + sc[t][c]; mu += v[c]; }
        mu *= (1.f / 32.f);
        float var = 0.f;
        for (int c = 0; c < 32; ++c) { float d = v[c] - mu; var += d * d; }
        var *= (1.f / 32.f);
        float inv = rsqrtf(var + LN_EPS);
        for (int c = 0; c < 32; ++c)
            xg[t * 32 + c] = (v[c] - mu) * inv * ln_g[c] + ln_b[c];
    }
}

// ---------------------------------------------------------------------------
// Kernel 3: column attention (seq length 4 over tracks). One block handles 16
// bins (=64 rows). Writes msa2 in (b, l, s, c) layout.
// ---------------------------------------------------------------------------
__global__ __launch_bounds__(256)
void k_col_attn(const float* __restrict__ msa, float* __restrict__ msa2,
                const float* __restrict__ w_in, const float* __restrict__ b_in,
                const float* __restrict__ w_out, const float* __restrict__ b_out,
                const float* __restrict__ ln_g, const float* __restrict__ ln_b) {
    __shared__ float  xs[64][32];
    __shared__ float  qkv[64][96];
    __shared__ float  os[64][32];
    __shared__ float  pr[64][32];
    __shared__ __bf16 win[96 * 32];
    __shared__ __bf16 wot[32 * 32];

    int t = threadIdx.x;
    int b = blockIdx.x >> 2, lch = blockIdx.x & 3;

    // row r = lo*4 + s  ->  x[r][c] = msa[b, s, lch*16+lo, c]
    for (int i = t; i < 64 * 32; i += 256) {
        int r = i >> 5, c = i & 31, lo = r >> 2, s = r & 3;
        xs[r][c] = msa[(((size_t)b * 4 + s) * 64 + lch * 16 + lo) * 32 + c];
    }
    for (int i = t; i < 96 * 32; i += 256) win[i] = (__bf16)w_in[i];
    for (int i = t; i < 32 * 32; i += 256) wot[i] = (__bf16)w_out[i];
    __syncthreads();

    int wave = t >> 5, lane = t & 31, g = lane >> 4, L = lane & 15;

    // ---- QKV projection (same tiling as row kernel) ----
    for (int q = 0; q < 3; ++q) {
        int tile = wave * 3 + q;
        int mt = tile / 6, nt = tile % 6;
        v16bf a, bm;
        #pragma unroll
        for (int e = 0; e < 16; ++e) {
            a[e]  = (__bf16)xs[mt * 16 + L][aK(e, g)];
            bm[e] = win[(nt * 16 + L) * 32 + (g << 4) + e];
        }
        float bias = b_in[nt * 16 + L];
        v8f acc;
        #pragma unroll
        for (int r = 0; r < 8; ++r) acc[r] = bias;
        acc = __builtin_amdgcn_wmma_f32_16x16x32_bf16(false, a, false, bm,
                                                      (short)0, acc, false, false);
        #pragma unroll
        for (int r = 0; r < 8; ++r) qkv[mt * 16 + r + 8 * g][nt * 16 + L] = acc[r];
    }
    __syncthreads();

    // ---- attention over 4 tracks, fully in registers: t -> (lo, h, i) ----
    {
        int lo = t >> 4, h = (t >> 2) & 3, i = t & 3;
        int base = lo * 4;
        float s0[4], mx = -1e30f;
        #pragma unroll
        for (int j = 0; j < 4; ++j) {
            float s = 0.f;
            #pragma unroll
            for (int dd = 0; dd < 8; ++dd)
                s += qkv[base + i][h * 8 + dd] * qkv[base + j][32 + h * 8 + dd];
            s0[j] = s * ATTN_SCALE;
            mx = fmaxf(mx, s0[j]);
        }
        float sum = 0.f;
        #pragma unroll
        for (int j = 0; j < 4; ++j) { s0[j] = __expf(s0[j] - mx); sum += s0[j]; }
        float inv = 1.f / sum;
        #pragma unroll
        for (int dd = 0; dd < 8; ++dd) {
            float o = 0.f;
            #pragma unroll
            for (int j = 0; j < 4; ++j) o += s0[j] * inv * qkv[base + j][64 + h * 8 + dd];
            os[base + i][h * 8 + dd] = o;
        }
    }
    __syncthreads();

    // ---- out projection into pr ----
    {
        int mt = wave >> 1, nt = wave & 1;
        v16bf a, bm;
        #pragma unroll
        for (int e = 0; e < 16; ++e) {
            a[e]  = (__bf16)os[mt * 16 + L][aK(e, g)];
            bm[e] = wot[(nt * 16 + L) * 32 + (g << 4) + e];
        }
        float bias = b_out[nt * 16 + L];
        v8f acc;
        #pragma unroll
        for (int r = 0; r < 8; ++r) acc[r] = bias;
        acc = __builtin_amdgcn_wmma_f32_16x16x32_bf16(false, a, false, bm,
                                                      (short)0, acc, false, false);
        #pragma unroll
        for (int r = 0; r < 8; ++r) pr[mt * 16 + r + 8 * g][nt * 16 + L] = acc[r];
    }
    __syncthreads();

    // ---- residual + LN, write msa2[b, l, s, c] ----
    if (t < 64) {
        float v[32], mu = 0.f;
        for (int c = 0; c < 32; ++c) { v[c] = xs[t][c] + pr[t][c]; mu += v[c]; }
        mu *= (1.f / 32.f);
        float var = 0.f;
        for (int c = 0; c < 32; ++c) { float d = v[c] - mu; var += d * d; }
        var *= (1.f / 32.f);
        float inv = rsqrtf(var + LN_EPS);
        int lo = t >> 2, s = t & 3, l = lch * 16 + lo;
        for (int c = 0; c < 32; ++c)
            msa2[(((size_t)b * 64 + l) * 4 + s) * 32 + c] =
                (v[c] - mu) * inv * ln_g[c] + ln_b[c];
    }
}

// ---------------------------------------------------------------------------
// Kernel 4: mean over tracks + L2 norm per (b,l). One wave per row.
// ---------------------------------------------------------------------------
__global__ __launch_bounds__(32)
void k_mean(const float* __restrict__ msa2, float* __restrict__ m,
            float* __restrict__ nrm) {
    int bl = blockIdx.x;       // b*64 + l, 0..4095
    int c = threadIdx.x;       // 0..31
    float s = 0.f;
    #pragma unroll
    for (int t = 0; t < 4; ++t) s += msa2[((size_t)bl * 4 + t) * 32 + c];
    float mv = s * 0.25f;
    m[bl * 32 + c] = mv;
    float sq = mv * mv;
    for (int k = 16; k > 0; k >>= 1) sq += __shfl_xor(sq, k, 32);
    if (c == 0) nrm[bl] = sqrtf(sq);
}

// ---------------------------------------------------------------------------
// Kernel 5: u[b,j,p,c] = sum_d proj_w[p, c*32+d] * m[b,j,d]
// GEMM (4096 x 32) @ (32 x 512), output stored bf16 as u[row][p*32+c].
// ---------------------------------------------------------------------------
__global__ __launch_bounds__(256)
void k_proj1(const float* __restrict__ m, const float* __restrict__ proj_w,
             __bf16* __restrict__ u) {
    __shared__ __bf16 pw[16 * 1024];
    int t = threadIdx.x;
    for (int i = t; i < 16 * 1024; i += 256) pw[i] = (__bf16)proj_w[i];
    __syncthreads();

    int wave = t >> 5, lane = t & 31, g = lane >> 4, L = lane & 15;
    int mt = blockIdx.x;                       // 256 row-tiles of 16
    v16bf a;
    #pragma unroll
    for (int e = 0; e < 16; ++e)
        a[e] = (__bf16)m[(size_t)(mt * 16 + L) * 32 + aK(e, g)];

    for (int nt = wave; nt < 32; nt += 8) {
        int n = nt * 16 + L;                   // n = p*32 + c
        int p = n >> 5, c = n & 31;
        v16bf bm;
        #pragma unroll
        for (int e = 0; e < 16; ++e)
            bm[e] = pw[p * 1024 + c * 32 + (g << 4) + e];   // K = d = 16g+e
        v8f acc = {};
        acc = __builtin_amdgcn_wmma_f32_16x16x32_bf16(false, a, false, bm,
                                                      (short)0, acc, false, false);
        #pragma unroll
        for (int r = 0; r < 8; ++r)
            u[(size_t)(mt * 16 + r + 8 * g) * 512 + n] = (__bf16)acc[r];
    }
}

// ---------------------------------------------------------------------------
// Kernel 6: feat = (m_i . u_j) / max(|m_i||m_j|,1e-6) + bias, LN over p=16,
// SiLU, scatter to out[b,p,i,j]. One block per (b, i-tile); CP=16 lies on the
// 16-lane half-wave of the D fragment -> LN via shfl_xor butterflies.
// ---------------------------------------------------------------------------
__global__ __launch_bounds__(256)
void k_proj2(const float* __restrict__ m, const float* __restrict__ nrm,
             const __bf16* __restrict__ u, const float* __restrict__ proj_b,
             const float* __restrict__ pg, const float* __restrict__ pb,
             float* __restrict__ out) {
    int t = threadIdx.x, wave = t >> 5, lane = t & 31, g = lane >> 4, L = lane & 15;
    int b = blockIdx.x >> 2, mt = blockIdx.x & 3;
    int i0 = b * 64 + mt * 16;

    v16bf a;
    #pragma unroll
    for (int e = 0; e < 16; ++e)
        a[e] = (__bf16)m[(size_t)(i0 + L) * 32 + aK(e, g)];

    float normi[8];
    #pragma unroll
    for (int r = 0; r < 8; ++r) normi[r] = nrm[i0 + r + 8 * g];

    int p = L;
    float pbias = proj_b[p], gam = pg[p], bet = pb[p];

    for (int j = wave; j < 64; j += 8) {
        // contiguous 32B per lane: u[b*64+j][p*32 + 16g .. +15]  (K = c = 16g+e)
        v16bf bm = *(const v16bf*)(u + ((size_t)(b * 64 + j)) * 512 + p * 32 + g * 16);
        v8f acc = {};
        acc = __builtin_amdgcn_wmma_f32_16x16x32_bf16(false, a, false, bm,
                                                      (short)0, acc, false, false);
        float normj = nrm[b * 64 + j];
        #pragma unroll
        for (int r = 0; r < 8; ++r) {
            float denom = fmaxf(normi[r] * normj, 1e-6f);
            float x = acc[r] / denom + pbias;
            // LN over the 16 p-values living in this 16-lane group
            float s1 = x, s2 = x * x;
            #pragma unroll
            for (int k = 8; k > 0; k >>= 1) {
                s1 += __shfl_xor(s1, k, 32);
                s2 += __shfl_xor(s2, k, 32);
            }
            float mu  = s1 * (1.f / 16.f);
            float var = s2 * (1.f / 16.f) - mu * mu;
            float y = (x - mu) * rsqrtf(var + LN_EPS) * gam + bet;
            float sv = y / (1.f + __expf(-y));          // SiLU
            int i = mt * 16 + r + 8 * g;
            out[(((size_t)b * 16 + p) * 64 + i) * 64 + j] = sv;
        }
    }
}

// ---------------------------------------------------------------------------
extern "C" void kernel_launch(void* const* d_in, const int* in_sizes, int n_in,
                              void* d_out, int out_size, void* d_ws, size_t ws_size,
                              hipStream_t stream) {
    const float* chip0   = (const float*)d_in[0];
    const float* chip1   = (const float*)d_in[1];
    const float* chip2   = (const float*)d_in[2];
    const float* chip3   = (const float*)d_in[3];
    const float* embed_w = (const float*)d_in[4];
    const float* embed_b = (const float*)d_in[5];
    const float* pos     = (const float*)d_in[6];
    const float* row_in_w  = (const float*)d_in[7];
    const float* row_in_b  = (const float*)d_in[8];
    const float* row_out_w = (const float*)d_in[9];
    const float* row_out_b = (const float*)d_in[10];
    const float* row_ln_g  = (const float*)d_in[11];
    const float* row_ln_b  = (const float*)d_in[12];
    const float* col_in_w  = (const float*)d_in[13];
    const float* col_in_b  = (const float*)d_in[14];
    const float* col_out_w = (const float*)d_in[15];
    const float* col_out_b = (const float*)d_in[16];
    const float* col_ln_g  = (const float*)d_in[17];
    const float* col_ln_b  = (const float*)d_in[18];
    const float* proj_w  = (const float*)d_in[19];
    const float* proj_b  = (const float*)d_in[20];
    const float* pln_g   = (const float*)d_in[21];
    const float* pln_b   = (const float*)d_in[22];

    char* ws = (char*)d_ws;
    float*  msa  = (float*)(ws);                 // 524288 f  (2 MB)
    float*  msa2 = (float*)(ws + 2097152);       // 524288 f  (2 MB)
    float*  m    = (float*)(ws + 4194304);       // 131072 f  (512 KB)
    float*  nrmv = (float*)(ws + 4718592);       // 4096 f    (16 KB)
    __bf16* u    = (__bf16*)(ws + 4734976);      // 2097152 bf16 (4 MB)
    float*  outp = (float*)d_out;

    k_embed<<<2048, 256, 0, stream>>>(chip0, chip1, chip2, chip3,
                                      embed_w, embed_b, pos, msa);
    k_row_attn<<<256, 256, 0, stream>>>(msa, row_in_w, row_in_b,
                                        row_out_w, row_out_b, row_ln_g, row_ln_b);
    k_col_attn<<<256, 256, 0, stream>>>(msa, msa2, col_in_w, col_in_b,
                                        col_out_w, col_out_b, col_ln_g, col_ln_b);
    k_mean<<<4096, 32, 0, stream>>>(msa2, m, nrmv);
    k_proj1<<<256, 256, 0, stream>>>(m, proj_w, u);
    k_proj2<<<256, 256, 0, stream>>>(m, nrmv, u, proj_b, pln_g, pln_b, outp);
}
